// SCAN_56186762166737
// MI455X (gfx1250) — matrix operand order repaired and checked by reading, not compiled, not run on hardware
//
#include <hip/hip_runtime.h>

typedef __attribute__((ext_vector_type(16))) __bf16       v16bf;
typedef __attribute__((ext_vector_type(8)))  float        v8f;
typedef __attribute__((ext_vector_type(4)))  unsigned int v4u;

#define N_IMG   64
#define N_CAP   64
#define N_REG   36
#define R_PAD   64
#define MAX_W   32
#define IMG_DIM 2048
#define DMODEL  1024
#define M_ROWS  (N_IMG * N_REG)   // 2304
#define CPG     4                 // captions per scan_sim block
#define PROJ_ROWS 32
#define EPSF    1e-8f
#define LAM_SM  9.0f
#define LAM_LSE 6.0f
#define LEAKY_K 0.1f

#ifndef __has_builtin
#define __has_builtin(x) 0
#endif
#if defined(__HIP_DEVICE_COMPILE__) && __has_builtin(__builtin_amdgcn_global_load_async_to_lds_b128)
#define HAVE_ASYNC_LDS 1
#else
#define HAVE_ASYNC_LDS 0
#endif

#if HAVE_ASYNC_LDS
typedef int v4i_vs __attribute__((vector_size(16)));
typedef __attribute__((address_space(1))) v4i_vs v4i_g;   // global (AS1) 128-bit chunk
typedef __attribute__((address_space(3))) v4i_vs v4i_l;   // LDS (AS3) 128-bit chunk
#endif

// ---------- helpers ----------
__device__ __forceinline__ unsigned short f2bf(float f) {
  unsigned int u = __float_as_uint(f);
  u += 0x7FFFu + ((u >> 16) & 1u);          // round-to-nearest-even
  return (unsigned short)(u >> 16);
}
__device__ __forceinline__ v8f v8f_zero() {
  v8f z;
#pragma unroll
  for (int i = 0; i < 8; ++i) z[i] = 0.0f;
  return z;
}
// Build a 16x bf16 fragment from two 16-byte chunks (A and B: +0/+8 elems).
__device__ __forceinline__ v16bf ldfrag(const unsigned short* p0, const unsigned short* p1) {
  union { v16bf v; v4u u[2]; } t;
  t.u[0] = *(const v4u*)p0;
  t.u[1] = *(const v4u*)p1;
  return t.v;
}
__device__ __forceinline__ v8f wmma_bf16(v16bf a, v16bf b, v8f c) {
  return __builtin_amdgcn_wmma_f32_16x16x32_bf16(false, a, false, b, (short)0, c, false, false);
}
__device__ __forceinline__ void async_wait0() {
#if HAVE_ASYNC_LDS
#if __has_builtin(__builtin_amdgcn_s_wait_asynccnt)
  __builtin_amdgcn_s_wait_asynccnt(0);
#else
  asm volatile("s_wait_asynccnt 0x0" ::: "memory");
#endif
#endif
}
// Copy `n16` 16-byte chunks global->LDS, distributed over 256 threads.
__device__ __forceinline__ void stage_to_lds(unsigned short* lds_dst,
                                             const unsigned short* gsrc,
                                             int n16, int tid) {
  for (int ch = tid; ch < n16; ch += 256) {
#if HAVE_ASYNC_LDS
    unsigned short* g = (unsigned short*)(gsrc + (size_t)ch * 8);   // drop const
    unsigned short* l = lds_dst + (size_t)ch * 8;
    __builtin_amdgcn_global_load_async_to_lds_b128((v4i_g*)g, (v4i_l*)l, 0, 0);
#else
    *(v4u*)(lds_dst + (size_t)ch * 8) = *(const v4u*)(gsrc + (size_t)ch * 8);
#endif
  }
}

// ---------- prep kernels ----------
__global__ void prep_wt(const float* __restrict__ W, unsigned short* __restrict__ Wt) {
  int idx = blockIdx.x * 256 + threadIdx.x;            // over DMODEL*IMG_DIM
  if (idx >= DMODEL * IMG_DIM) return;
  int n = idx / IMG_DIM, k = idx - n * IMG_DIM;
  Wt[idx] = f2bf(W[(size_t)k * DMODEL + n]);           // Wt[n][k] = W[k][n]
}

__global__ void prep_img(const float* __restrict__ im, unsigned short* __restrict__ o) {
  int idx = blockIdx.x * 256 + threadIdx.x;
  if (idx < M_ROWS * IMG_DIM) o[idx] = f2bf(im[idx]);
}

__global__ void zero16(v4u* __restrict__ p, int n) {
  int idx = blockIdx.x * 256 + threadIdx.x;
  if (idx < n) { v4u z = {0u, 0u, 0u, 0u}; p[idx] = z; }
}

// ---------- caption l2norm ----------
__global__ __launch_bounds__(256) void cap_norm(const float* __restrict__ cap,
                                                unsigned short* __restrict__ capb,
                                                float* __restrict__ w1) {
  __shared__ float red[256];
  const float* row = cap + (size_t)blockIdx.x * DMODEL;
  float v[4]; float s = 0.f;
#pragma unroll
  for (int j = 0; j < 4; ++j) { v[j] = row[threadIdx.x + j * 256]; s += v[j] * v[j]; }
  red[threadIdx.x] = s;
  __syncthreads();
  for (int st = 128; st > 0; st >>= 1) {
    if (threadIdx.x < st) red[threadIdx.x] += red[threadIdx.x + st];
    __syncthreads();
  }
  float nrm = sqrtf(red[0]);
  float inv = 1.0f / (nrm + EPSF);
  unsigned short* ob = capb + (size_t)blockIdx.x * DMODEL;
#pragma unroll
  for (int j = 0; j < 4; ++j) ob[threadIdx.x + j * 256] = f2bf(v[j] * inv);
  if (threadIdx.x == 0) w1[blockIdx.x] = nrm * inv;    // |l2norm(x)| = n/(n+eps)
}

// ---------- image projection GEMM + row l2norm ----------
// grid = 72 blocks (32 rows each), 512 threads = 16 waves
// wave: mtile = wid&1 (16 rows), colgroup = wid>>1 (128 cols)
__global__ __launch_bounds__(512) void img_proj(const unsigned short* __restrict__ Abf,
                                                const unsigned short* __restrict__ Wt,
                                                const float* __restrict__ bias,
                                                unsigned short* __restrict__ Ebf,
                                                unsigned short* __restrict__ Et) {
  __shared__ float rowss[PROJ_ROWS];
  __shared__ float invn[PROJ_ROWS];
  int tid = threadIdx.x;
  int wid = tid >> 5, lane = tid & 31;
  int lrow = lane & 15, lhalf = lane >> 4;
  int mtile = wid & 1, cgp = wid >> 1;
  if (tid < PROJ_ROWS) rowss[tid] = 0.f;
  __syncthreads();

  int rowblk = blockIdx.x * PROJ_ROWS;
  v8f acc[8];
#pragma unroll
  for (int s = 0; s < 8; ++s) acc[s] = v8f_zero();

  const unsigned short* Arow = Abf + (size_t)(rowblk + mtile * 16 + lrow) * IMG_DIM;
  for (int k = 0; k < IMG_DIM; k += 32) {
    int ka = k + lhalf * 16;
    v16bf a = ldfrag(Arow + ka, Arow + ka + 8);
    int kb = k + lhalf * 16;
#pragma unroll
    for (int s = 0; s < 8; ++s) {
      const unsigned short* Bp = Wt + (size_t)(cgp * 128 + s * 16 + lrow) * IMG_DIM + kb;
      v16bf b = ldfrag(Bp, Bp + 8);
      acc[s] = wmma_bf16(a, b, acc[s]);
    }
  }

  // bias + per-row sum of squares
  float ss[8];
#pragma unroll
  for (int j = 0; j < 8; ++j) ss[j] = 0.f;
#pragma unroll
  for (int s = 0; s < 8; ++s) {
    int col = cgp * 128 + s * 16 + lrow;
    float bv = bias[col];
#pragma unroll
    for (int j = 0; j < 8; ++j) {
      float v = acc[s][j] + bv;
      acc[s][j] = v;
      ss[j] += v * v;
    }
  }
#pragma unroll
  for (int j = 0; j < 8; ++j) atomicAdd(&rowss[mtile * 16 + lhalf * 8 + j], ss[j]);
  __syncthreads();
  if (tid < PROJ_ROWS) invn[tid] = 1.0f / (sqrtf(rowss[tid]) + EPSF);
  __syncthreads();

#pragma unroll
  for (int j = 0; j < 8; ++j) {
    int row32 = mtile * 16 + lhalf * 8 + j;
    int grow = rowblk + row32;
    float inv = invn[row32];
    int ii = grow / N_REG;
    int rr = grow - ii * N_REG;
#pragma unroll
    for (int s = 0; s < 8; ++s) {
      int col = cgp * 128 + s * 16 + lrow;
      unsigned short hv = f2bf(acc[s][j] * inv);
      Ebf[(size_t)grow * DMODEL + col] = hv;
      Et[((size_t)ii * DMODEL + col) * R_PAD + rr] = hv;   // transposed, r-padded
    }
  }
}

// ---------- fused attention + softmax + |wctx| + LSE ----------
// grid = 1024 blocks: cg = bid & 15 (4 captions), i = bid >> 4; 256 threads = 8 waves
__global__ __launch_bounds__(256) void scan_sim(const unsigned short* __restrict__ Ebf,
                                                const unsigned short* __restrict__ Et,
                                                const unsigned short* __restrict__ capb,
                                                const float* __restrict__ w1,
                                                const int* __restrict__ lens,
                                                float* __restrict__ out) {
  const int HALF = DMODEL / 2;                       // 512 d-cols per staging pass
  __shared__ unsigned short EtS[HALF * R_PAD];       // 64 KB staged Et^T half
  __shared__ float Sraw[CPG][R_PAD][MAX_W];          // 32 KB raw dots (for w12)
  __shared__ float Sm[CPG][R_PAD][MAX_W];            // 32 KB masked/normed scores
  __shared__ unsigned short Asm[CPG][MAX_W][R_PAD];  // 16 KB softmaxed attn (bf16)
  __shared__ float w12v[CPG][MAX_W];
  __shared__ float w2s[CPG][MAX_W];

  int cg = blockIdx.x & 15;
  int i  = blockIdx.x >> 4;
  int tid = threadIdx.x;
  int wid = tid >> 5, lane = tid & 31;
  int lrow = lane & 15, lhalf = lane >> 4;
  if (tid < CPG * MAX_W) ((float*)w2s)[tid] = 0.f;

  const unsigned short* EtG = Et + (size_t)i * DMODEL * R_PAD;
  // stage first half of Et^T[i] (overlaps with phase 1 compute)
  stage_to_lds(EtS, EtG, (HALF * R_PAD) / 8, tid);

  // ---- phase 1: Sraw[cc][r][w] = img_emb[i,r,:] . cap[c,w,:]  (K = 1024)
  int rtile = wid >> 1, wtile = wid & 1;
  int r_a = rtile * 16 + lrow;
  int r_cl = (r_a < N_REG) ? r_a : (N_REG - 1);      // clamp padded rows (masked later)
  const unsigned short* Ar = Ebf + ((size_t)i * N_REG + r_cl) * DMODEL;
  const unsigned short* Bc[CPG];
  int lenv[CPG];
#pragma unroll
  for (int cc = 0; cc < CPG; ++cc) {
    int c = cg * CPG + cc;
    Bc[cc] = capb + ((size_t)c * MAX_W + wtile * 16 + lrow) * DMODEL;
    lenv[cc] = lens[c];
  }
  v8f acc[CPG];
#pragma unroll
  for (int cc = 0; cc < CPG; ++cc) acc[cc] = v8f_zero();
  for (int k = 0; k < DMODEL; k += 32) {
    v16bf a = ldfrag(Ar + k + lhalf * 16, Ar + k + lhalf * 16 + 8);
#pragma unroll
    for (int cc = 0; cc < CPG; ++cc) {
      v16bf b = ldfrag(Bc[cc] + k + lhalf * 16, Bc[cc] + k + lhalf * 16 + 8);
      acc[cc] = wmma_bf16(a, b, acc[cc]);
    }
  }
  int wcol = wtile * 16 + lrow;
#pragma unroll
  for (int cc = 0; cc < CPG; ++cc) {
    bool wvalid = (wcol < lenv[cc]);
#pragma unroll
    for (int j = 0; j < 8; ++j) {
      int r = rtile * 16 + lhalf * 8 + j;
      float v = acc[cc][j];
      Sraw[cc][r][wcol] = v;
      float lv = (v > 0.f) ? v : LEAKY_K * v;        // LeakyReLU(0.1)
      if (!wvalid || r >= N_REG) lv = 0.f;           // word mask + r padding
      Sm[cc][r][wcol] = lv;
    }
  }
  __syncthreads();

  // ---- per-r l2norm over w, scaled by lambda_softmax (256 threads = 4c x 64r)
  {
    int cc = tid >> 6, r = tid & 63;
    float ssq = 0.f;
#pragma unroll
    for (int w = 0; w < MAX_W; ++w) ssq += Sm[cc][r][w] * Sm[cc][r][w];
    float sc = LAM_SM / (sqrtf(ssq) + EPSF);
#pragma unroll
    for (int w = 0; w < MAX_W; ++w) Sm[cc][r][w] *= sc;
  }
  __syncthreads();

  // ---- softmax over r per (cc,w) + w12 = sum_r attn * Sraw
  if (tid < CPG * MAX_W) {
    int cc = tid >> 5, w = tid & 31;
    float mx = -1e30f;
    for (int r = 0; r < N_REG; ++r) mx = fmaxf(mx, Sm[cc][r][w]);
    float sum = 0.f;
    for (int r = 0; r < N_REG; ++r) sum += __expf(Sm[cc][r][w] - mx);
    float inv = 1.0f / sum;
    float w12 = 0.f;
    for (int r = 0; r < N_REG; ++r) {
      float aa = __expf(Sm[cc][r][w] - mx) * inv;
      Asm[cc][w][r] = f2bf(aa);
      w12 += aa * Sraw[cc][r][w];                    // cap . wctx, exactly
    }
    for (int r = N_REG; r < R_PAD; ++r) Asm[cc][w][r] = 0;
    w12v[cc][w] = w12;
  }
  async_wait0();
  __syncthreads();

  // ---- phase 2: |wctx|^2 via wctx tiles (M=32 w, K=64 rpad), two d-halves
  int mtile = wid & 1;
  int cb = (wid >> 1) * 128;                         // local d-col base within half
  int wrow = mtile * 16 + lhalf * 8;
  float p2[CPG][8];
#pragma unroll
  for (int cc = 0; cc < CPG; ++cc)
#pragma unroll
    for (int j = 0; j < 8; ++j) p2[cc][j] = 0.f;

  for (int half = 0; half < 2; ++half) {
    if (half == 1) {
      __syncthreads();                               // everyone done reading half 0
      stage_to_lds(EtS, EtG + (size_t)HALF * R_PAD, (HALF * R_PAD) / 8, tid);
      async_wait0();
      __syncthreads();
    }
    for (int s = 0; s < 8; ++s) {
      int dloc = cb + s * 16 + lrow;
      const unsigned short* Bp = &EtS[dloc * R_PAD];
      v16bf b0 = ldfrag(Bp + lhalf * 16, Bp + lhalf * 16 + 8);
      v16bf b1 = ldfrag(Bp + 32 + lhalf * 16, Bp + 32 + lhalf * 16 + 8);
#pragma unroll
      for (int cc = 0; cc < CPG; ++cc) {
        const unsigned short* Ap = &Asm[cc][mtile * 16 + lrow][0];
        v16bf a0 = ldfrag(Ap + lhalf * 16, Ap + lhalf * 16 + 8);
        v16bf a1 = ldfrag(Ap + 32 + lhalf * 16, Ap + 32 + lhalf * 16 + 8);
        v8f a2 = v8f_zero();
        a2 = wmma_bf16(a0, b0, a2);
        a2 = wmma_bf16(a1, b1, a2);
#pragma unroll
        for (int j = 0; j < 8; ++j) p2[cc][j] += a2[j] * a2[j];
      }
    }
  }
#pragma unroll
  for (int cc = 0; cc < CPG; ++cc)
#pragma unroll
    for (int j = 0; j < 8; ++j) atomicAdd(&w2s[cc][wrow + j], p2[cc][j]);
  __syncthreads();

  // ---- phase 3: cosine + masked LSE over words (4 waves, one per caption)
  if (tid < CPG * MAX_W) {
    int cc = tid >> 5, w = tid & 31;
    int c = cg * CPG + cc;
    float w2v = sqrtf(w2s[cc][w]);
    float w1v = w1[c * MAX_W + w];
    float rs = w12v[cc][w] / fmaxf(w1v * w2v, EPSF);
    float e = (w < lens[c]) ? __expf(rs * LAM_LSE) : 0.f;
#pragma unroll
    for (int m = 16; m >= 1; m >>= 1) e += __shfl_xor(e, m, 32);
    if (w == 0) out[(size_t)i * N_CAP + c] = __logf(e) / LAM_LSE;   // sim.T
  }
}

// ---------- launch ----------
extern "C" void kernel_launch(void* const* d_in, const int* in_sizes, int n_in,
                              void* d_out, int out_size, void* d_ws, size_t ws_size,
                              hipStream_t stream) {
  const float* images  = (const float*)d_in[0];
  const float* cap_emb = (const float*)d_in[1];
  const float* W_fc    = (const float*)d_in[2];
  const float* b_fc    = (const float*)d_in[3];
  const int*   lens    = (const int*)d_in[4];
  float* out = (float*)d_out;

  char* ws = (char*)d_ws;
  size_t off = 0;
  auto take = [&](size_t bytes) -> char* {
    char* p = ws + off;
    off = (off + bytes + 255) & ~(size_t)255;
    return p;
  };
  unsigned short* Wt   = (unsigned short*)take((size_t)DMODEL * IMG_DIM * 2);
  unsigned short* Ibf  = (unsigned short*)take((size_t)M_ROWS * IMG_DIM * 2);
  unsigned short* Capb = (unsigned short*)take((size_t)N_CAP * MAX_W * DMODEL * 2);
  float*          W1   = (float*)take((size_t)N_CAP * MAX_W * 4);
  unsigned short* Ebf  = (unsigned short*)take((size_t)M_ROWS * DMODEL * 2);
  unsigned short* Et   = (unsigned short*)take((size_t)N_IMG * DMODEL * R_PAD * 2);
  (void)ws_size; (void)in_sizes; (void)n_in; (void)out_size;

  int nWt = DMODEL * IMG_DIM;
  prep_wt<<<(nWt + 255) / 256, 256, 0, stream>>>(W_fc, Wt);
  int nIm = M_ROWS * IMG_DIM;
  prep_img<<<(nIm + 255) / 256, 256, 0, stream>>>(images, Ibf);
  int nEt16 = (N_IMG * DMODEL * R_PAD * 2) / 16;
  zero16<<<(nEt16 + 255) / 256, 256, 0, stream>>>((v4u*)Et, nEt16);

  cap_norm<<<N_CAP * MAX_W, 256, 0, stream>>>(cap_emb, Capb, W1);
  img_proj<<<M_ROWS / PROJ_ROWS, 512, 0, stream>>>(Ibf, Wt, b_fc, Ebf, Et);
  scan_sim<<<(N_CAP / CPG) * N_IMG, 256, 0, stream>>>(Ebf, Et, Capb, W1, lens, out);
}
